// MultiHeadAttention_46995532153577
// MI455X (gfx1250) — matrix operand order repaired and checked by reading, not compiled, not run on hardware
//
#include <hip/hip_runtime.h>
#include <hip/hip_bf16.h>

// MultiHeadAttention, MI455X (gfx1250): f16 WMMA (16x16x32, f32 acc) for all
// GEMMs + flash-attention with fp32 online softmax.
// B=2, S=2048, D=1024, H=16, DK=64.

#define B_   2
#define S_   2048
#define D_   1024
#define H_   16
#define DK_  64
#define M_   (B_ * S_)   // 4096 tokens

typedef _Float16 f16;
typedef __attribute__((ext_vector_type(16))) _Float16 v16h;
typedef __attribute__((ext_vector_type(8)))  _Float16 v8h;
typedef __attribute__((ext_vector_type(8)))  float    v8f;

static __device__ __forceinline__ v8f wmma_f16(v16h a, v16h b, v8f c) {
  // D = A(16x32 f16) * B(32x16 f16) + C(16x16 f32)
  return __builtin_amdgcn_wmma_f32_16x16x32_f16(false, a, false, b, (short)0, c,
                                                false, false);
}

// A-fragment (16x32 f16): lane L (half = L>>4, r = L&15) holds
// row r, K = half*16 .. half*16+15 (contiguous, same order as B-fragment).
static __device__ __forceinline__ v16h load_a_frag(const f16* __restrict__ rowPtr,
                                                   int kBase, int half) {
  return *(const v16h*)(rowPtr + kBase + half * 16);
}
// B-fragment (32x16 f16): lane L holds column n=L&15, K = half*16 .. half*16+15
// (16 contiguous f16 = one 32B load when the K-major row is contiguous).

__global__ void cvt_f32_to_f16(const float* __restrict__ in,
                               f16* __restrict__ out, int n) {
  int i = blockIdx.x * blockDim.x + threadIdx.x;
  if (i < n) out[i] = (f16)in[i];
}

// out = A[4096x1024](f16, row-major) @ W^T + bias, W[j][k] row-major (j = out col).
// mode 0: f16 head-major   [b][h][s][dk], scaled by `scale` (q gets 1/sqrt(DK))
// mode 1: f16 head-major-T [b][h][dk][s]  (v, so P@V B-operand is K-contiguous)
// mode 2: f32 row-major    [m][col]       (final output)
__global__ __launch_bounds__(256) void gemm_xwT(
    const f16* __restrict__ A, const f16* __restrict__ W,
    const float* __restrict__ bias, void* __restrict__ out,
    int mode, float scale)
{
  const int lane = threadIdx.x & 31;
  const int wave = threadIdx.x >> 5;
  const int half = lane >> 4;
  const int nIn  = lane & 15;
  const int row0 = blockIdx.x * 128 + wave * 16;   // 8 waves stacked along M
  const int col0 = blockIdx.y * 64;                // 4 N-tiles per wave

  const f16* aRow = A + (size_t)(row0 + nIn) * D_;
  v8f acc[4] = {};

  for (int k0 = 0; k0 < D_; k0 += 32) {
    v16h a = load_a_frag(aRow, k0, half);
#pragma unroll
    for (int nt = 0; nt < 4; ++nt) {
      const f16* wRow = W + (size_t)(col0 + nt * 16 + nIn) * D_;
      v16h b = *(const v16h*)(wRow + k0 + half * 16);
      acc[nt] = wmma_f16(a, b, acc[nt]);
    }
  }

  // C layout: reg i -> row m = i + 8*half, col n = lane&15
#pragma unroll
  for (int nt = 0; nt < 4; ++nt) {
    const int col = col0 + nt * 16 + nIn;
    const float bb = bias[col];
#pragma unroll
    for (int i = 0; i < 8; ++i) {
      const int m = row0 + i + 8 * half;
      const float v = acc[nt][i] + bb;
      if (mode == 2) {
        ((float*)out)[(size_t)m * D_ + col] = v;
      } else {
        const int b = m >> 11, s = m & (S_ - 1);
        const int h = col >> 6, dk = col & (DK_ - 1);
        if (mode == 0)
          ((f16*)out)[((size_t)(b * H_ + h) * S_ + s) * DK_ + dk] = (f16)(v * scale);
        else
          ((f16*)out)[((size_t)(b * H_ + h) * DK_ + dk) * S_ + s] = (f16)v;
      }
    }
  }
}

// One wave per 16-query tile; 32 keys per inner iteration (8 WMMAs/iter).
__global__ __launch_bounds__(128) void flash_attn(
    const f16* __restrict__ qws,   // [B*H][S][DK], pre-scaled by 1/sqrt(DK)
    const f16* __restrict__ kws,   // [B*H][S][DK]
    const f16* __restrict__ vtws,  // [B*H][DK][S]
    f16* __restrict__ attn)        // [B][S][D]  (token-major for final GEMM)
{
  __shared__ __align__(64) f16 Ptile[4][16 * 32];   // per-wave P staging (C->A relayout)

  const int lane = threadIdx.x & 31;
  const int wave = threadIdx.x >> 5;
  const int half = lane >> 4;
  const int nIn  = lane & 15;

  const int wid   = blockIdx.x * 4 + wave;
  const int qt    = wid & (S_ / 16 - 1);
  const int bh    = wid >> 7;           // S_/16 == 128
  const int qBase = qt * 16;

  const f16* qb = qws  + (size_t)bh * S_  * DK_;
  const f16* kb = kws  + (size_t)bh * S_  * DK_;
  const f16* vb = vtws + (size_t)bh * DK_ * S_;
  f16* Pw = &Ptile[wave][0];

  // Q A-fragments, contraction dim = dk (64 -> two 32-wide steps)
  const f16* qRow = qb + (size_t)(qBase + nIn) * DK_;
  v16h a0 = load_a_frag(qRow, 0, half);
  v16h a1 = load_a_frag(qRow, 32, half);

  v8f O[4] = {};
  float mrun[8], lrun[8];
#pragma unroll
  for (int i = 0; i < 8; ++i) { mrun[i] = -1e30f; lrun[i] = 0.f; }

  const int ktMax = (qBase + 15) >> 5;          // causal: last key = qBase+15
  for (int kt = 0; kt <= ktMax; ++kt) {
    const int kBase = kt * 32;

    // S[16x32] = q @ k^T : B columns are k rows (dk-contiguous)
    v8f s0 = {}, s1 = {};
    {
      const f16* k0r = kb + (size_t)(kBase + nIn) * DK_ + half * 16;
      const f16* k1r = k0r + 16 * DK_;
      v16h b00 = *(const v16h*)(k0r);
      v16h b01 = *(const v16h*)(k0r + 32);
      v16h b10 = *(const v16h*)(k1r);
      v16h b11 = *(const v16h*)(k1r + 32);
      s0 = wmma_f16(a0, b00, s0);
      s0 = wmma_f16(a1, b01, s0);
      s1 = wmma_f16(a0, b10, s1);
      s1 = wmma_f16(a1, b11, s1);
    }

    const bool needMask = (kBase + 31 > qBase);   // wave-uniform

    // Online softmax. Row m = i + 8*half lives in the 16 lanes of one half,
    // so xor-shuffles with mask<16 are exact per-row reductions + broadcast.
#pragma unroll
    for (int i = 0; i < 8; ++i) {
      const int qrow = qBase + i + 8 * half;
      float e0 = s0[i], e1 = s1[i];
      if (needMask) {
        if (kBase + nIn > qrow)      e0 = -1e30f;   // mask==0 -> -inf score
        if (kBase + 16 + nIn > qrow) e1 = -1e30f;
      }
      float tmax = fmaxf(e0, e1);
#pragma unroll
      for (int d = 1; d < 16; d <<= 1) tmax = fmaxf(tmax, __shfl_xor(tmax, d, 32));
      const float mnew = fmaxf(mrun[i], tmax);
      const float sc = __expf(mrun[i] - mnew);
      const float p0 = __expf(e0 - mnew);
      const float p1 = __expf(e1 - mnew);
      float rsum = p0 + p1;
#pragma unroll
      for (int d = 1; d < 16; d <<= 1) rsum += __shfl_xor(rsum, d, 32);
      lrun[i] = lrun[i] * sc + rsum;
      mrun[i] = mnew;
#pragma unroll
      for (int nt = 0; nt < 4; ++nt) O[nt][i] *= sc;   // rescale prior accum
      const int r = i + 8 * half;
      Pw[r * 32 + nIn]      = (f16)p0;
      Pw[r * 32 + 16 + nIn] = (f16)p1;
    }

    // O[16x64] += P[16x32] @ V[32x64]; B columns (dk) are K-contiguous in v^T
    v16h pa = load_a_frag(Pw + nIn * 32, 0, half);
#pragma unroll
    for (int nt = 0; nt < 4; ++nt) {
      const int dk = nt * 16 + nIn;
      v16h bv = *(const v16h*)(vb + (size_t)dk * S_ + kBase + half * 16);
      O[nt] = wmma_f16(pa, bv, O[nt]);
    }
  }

  // normalize (reference: e / (sum + 1e-9)) and store token-major f16
  const int h = bh & (H_ - 1);
  const int b = bh >> 4;
  float inv[8];
#pragma unroll
  for (int i = 0; i < 8; ++i) inv[i] = 1.0f / (lrun[i] + 1e-9f);
#pragma unroll
  for (int nt = 0; nt < 4; ++nt) {
#pragma unroll
    for (int i = 0; i < 8; ++i) {
      const int s = qBase + i + 8 * half;
      const int d = h * DK_ + nt * 16 + nIn;
      attn[((size_t)(b * S_ + s)) * D_ + d] = (f16)(O[nt][i] * inv[i]);
    }
  }
}

extern "C" void kernel_launch(void* const* d_in, const int* in_sizes, int n_in,
                              void* d_out, int out_size, void* d_ws, size_t ws_size,
                              hipStream_t stream) {
  const float* Q  = (const float*)d_in[0];
  const float* K  = (const float*)d_in[1];
  const float* V  = (const float*)d_in[2];
  // d_in[3] = causal mask (unused: mask applied analytically)
  const float* Wq = (const float*)d_in[4];
  const float* bq = (const float*)d_in[5];
  const float* Wk = (const float*)d_in[6];
  const float* bk = (const float*)d_in[7];
  const float* Wv = (const float*)d_in[8];
  const float* bv = (const float*)d_in[9];
  const float* Wo = (const float*)d_in[10];
  const float* bo = (const float*)d_in[11];

  const size_t NMK = (size_t)M_ * D_;   // 4M elems
  const size_t NW  = (size_t)D_ * D_;   // 1M elems
  f16* p = (f16*)d_ws;                  // total ws use: 32M f16 = 64 MB
  f16* Qh    = p; p += NMK;
  f16* Kh    = p; p += NMK;
  f16* Vh    = p; p += NMK;
  f16* Wqh   = p; p += NW;
  f16* Wkh   = p; p += NW;
  f16* Wvh   = p; p += NW;
  f16* Woh   = p; p += NW;
  f16* qws   = p; p += NMK;
  f16* kws   = p; p += NMK;
  f16* vtws  = p; p += NMK;
  f16* attnh = p; p += NMK;

  const int CB = 256;
  cvt_f32_to_f16<<<(int)(NMK / CB), CB, 0, stream>>>(Q,  Qh,  (int)NMK);
  cvt_f32_to_f16<<<(int)(NMK / CB), CB, 0, stream>>>(K,  Kh,  (int)NMK);
  cvt_f32_to_f16<<<(int)(NMK / CB), CB, 0, stream>>>(V,  Vh,  (int)NMK);
  cvt_f32_to_f16<<<(int)(NW  / CB), CB, 0, stream>>>(Wq, Wqh, (int)NW);
  cvt_f32_to_f16<<<(int)(NW  / CB), CB, 0, stream>>>(Wk, Wkh, (int)NW);
  cvt_f32_to_f16<<<(int)(NW  / CB), CB, 0, stream>>>(Wv, Wvh, (int)NW);
  cvt_f32_to_f16<<<(int)(NW  / CB), CB, 0, stream>>>(Wo, Woh, (int)NW);

  dim3 gGrid(M_ / 128, D_ / 64);
  gemm_xwT<<<gGrid, 256, 0, stream>>>(Qh, Wqh, bq, qws, 0, 0.125f); // q * 1/sqrt(64)
  gemm_xwT<<<gGrid, 256, 0, stream>>>(Kh, Wkh, bk, kws, 0, 1.0f);
  gemm_xwT<<<gGrid, 256, 0, stream>>>(Vh, Wvh, bv, vtws, 1, 1.0f);  // transposed v

  flash_attn<<<(B_ * H_ * (S_ / 16)) / 4, 128, 0, stream>>>(qws, kws, vtws, attnh);

  gemm_xwT<<<gGrid, 256, 0, stream>>>(attnh, Woh, bo, d_out, 2, 1.0f);
}
